// MambaTemporal_11630771437656
// MI455X (gfx1250) — matrix operand order repaired and checked by reading, not compiled, not run on hardware
//
#include <hip/hip_runtime.h>
#include <hip/hip_bf16.h>

// ---------------------------------------------------------------------------
// Mamba temporal block on MI455X (gfx1250), bf16 WMMA GEMMs + fused scan.
// B=4, L=512, N_IN=1408, N_OUT=72, D_MODEL=1024, D_INNER=2048, D_STATE=16,
// D_CONV=4, DT_RANK=64, NL=2.  All GEMMs use v_wmma_f32_16x16x32_bf16.
// N-edge tiles are handled by CLAMPING weight-row loads (always in-bounds,
// always b128) and guarding only the stores -> single branch-free hot loop.
// ---------------------------------------------------------------------------

#define BATCH   4
#define SEQL    512
#define NIN     1408
#define NOUT    72
#define DMODEL  1024
#define DSTATE  16
#define DCONV   4
#define NL      2
#define DINNER  2048
#define DTRANK  64
#define ROWS    (BATCH * SEQL)   // 2048

typedef __attribute__((ext_vector_type(16))) __bf16 v16bf;
typedef __attribute__((ext_vector_type(8)))  float  v8f;

enum { EPI_NONE = 0, EPI_BIAS_GELU = 1, EPI_BIAS_SOFTPLUS = 2, EPI_RES = 3, EPI_BIAS = 4 };

// --------------------------- WMMA GEMM ------------------------------------
// C[M,N] = epilogue( A[M,K] * W[N,K]^T )   (A,W fp32 in memory -> bf16 frags)
// One wave computes a 32x32 tile (2x2 of 16x16 WMMA tiles).

template<int EPI>
__device__ inline void store_col8(const v8f& c, int mrow, int col, int N,
                                  const float* __restrict__ bias,
                                  const float* __restrict__ res,
                                  float* __restrict__ C, int ldc)
{
    if (col >= N) return;                 // only stores are guarded
    float bv = 0.0f;
    if (EPI == EPI_BIAS_GELU || EPI == EPI_BIAS_SOFTPLUS || EPI == EPI_BIAS)
        bv = bias[col];
#pragma unroll
    for (int r = 0; r < 8; ++r) {
        float v = c[r] + bv;
        if (EPI == EPI_BIAS_GELU) {
            v = 0.5f * v * (1.0f + erff(v * 0.70710678118654752f));   // exact GELU
        } else if (EPI == EPI_BIAS_SOFTPLUS) {
            v = (v > 20.0f) ? v : log1pf(__expf(v));                  // softplus
        } else if (EPI == EPI_RES) {
            v = c[r] + res[(size_t)(mrow + r) * ldc + col];
        }
        C[(size_t)(mrow + r) * ldc + col] = v;
    }
}

template<int EPI>
__global__ __launch_bounds__(256)
void gemm_bf16_wmma(const float* __restrict__ A, int lda,
                    const float* __restrict__ W, int ldw,
                    const float* __restrict__ bias,
                    const float* __restrict__ res,
                    float* __restrict__ C, int ldc,
                    int M, int N, int K)
{
    const int lane = threadIdx.x & 31;
    const int TN = (N + 31) >> 5;
    const int TM = M >> 5;
    const int wave = blockIdx.x * 8 + (threadIdx.x >> 5);
    if (wave >= TM * TN) return;          // uniform per wave: EXEC stays all-1s
    const int m0 = (wave / TN) * 32;
    const int n0 = (wave % TN) * 32;

    // A fragment addressing (16x32 bf16, contiguous K order shared with B):
    //   lane 0-15 -> M=lane, lane 16-31 -> same M, upper-K half.
    const int half = lane >> 4;
    const float* a0p = A + (size_t)(m0 + (lane & 15)) * lda;
    const float* a1p = A + (size_t)(m0 + 16 + (lane & 15)) * lda;

    // B fragment: B[k][n] = W[n][k]; lane -> column n, 16 contiguous K.
    // Clamp rows to N-1: loads always in-bounds; garbage columns never stored.
    const int nB0 = min(n0 + (lane & 15), N - 1);
    const int nB1 = min(n0 + 16 + (lane & 15), N - 1);
    const float* b0p = W + (size_t)nB0 * ldw;
    const float* b1p = W + (size_t)nB1 * ldw;

    v8f c00 = {}, c01 = {}, c10 = {}, c11 = {};

#pragma unroll 2
    for (int k = 0; k < K; k += 32) {
        v16bf a0, a1, b0, b1;
        const int ka   = k + 16 * half;   // A chunk K = 16h + [0..15]
        const int kb16 = k + 16 * half;   // B chunk K = 16h + [0..15]
#pragma unroll
        for (int i = 0; i < 8; ++i) {
            a0[i]     = (__bf16)a0p[ka + i];
            a0[8 + i] = (__bf16)a0p[ka + 8 + i];
            a1[i]     = (__bf16)a1p[ka + i];
            a1[8 + i] = (__bf16)a1p[ka + 8 + i];
        }
#pragma unroll
        for (int e = 0; e < 16; ++e) {
            b0[e] = (__bf16)b0p[kb16 + e];
            b1[e] = (__bf16)b1p[kb16 + e];
        }
        if (k + 32 < K) {   // global_prefetch_b8 of next K-slab
            __builtin_prefetch(a0p + k + 32, 0, 0);
            __builtin_prefetch(a1p + k + 32, 0, 0);
            __builtin_prefetch(b0p + k + 32, 0, 0);
            __builtin_prefetch(b1p + k + 32, 0, 0);
        }
        c00 = __builtin_amdgcn_wmma_f32_16x16x32_bf16(false, a0, false, b0, (short)0, c00, false, false);
        c01 = __builtin_amdgcn_wmma_f32_16x16x32_bf16(false, a0, false, b1, (short)0, c01, false, false);
        c10 = __builtin_amdgcn_wmma_f32_16x16x32_bf16(false, a1, false, b0, (short)0, c10, false, false);
        c11 = __builtin_amdgcn_wmma_f32_16x16x32_bf16(false, a1, false, b1, (short)0, c11, false, false);
    }

    // C/D layout: lanes 0-15 -> N=lane, rows M=r; lanes 16-31 -> rows M=8+r.
    const int colL  = lane & 15;
    const int rbase = (lane >> 4) * 8;
    store_col8<EPI>(c00, m0 + rbase,      n0 + colL,      N, bias, res, C, ldc);
    store_col8<EPI>(c01, m0 + rbase,      n0 + 16 + colL, N, bias, res, C, ldc);
    store_col8<EPI>(c10, m0 + 16 + rbase, n0 + colL,      N, bias, res, C, ldc);
    store_col8<EPI>(c11, m0 + 16 + rbase, n0 + 16 + colL, N, bias, res, C, ldc);
}

// --------------------------- LayerNorm ------------------------------------
__global__ __launch_bounds__(256)
void layernorm_k(const float* __restrict__ X, const float* __restrict__ g,
                 const float* __restrict__ b, float* __restrict__ Y, int D)
{
    __shared__ float sm[8], sv[8];
    const int row = blockIdx.x;
    const float* x = X + (size_t)row * D;
    float s = 0.0f, s2 = 0.0f;
    for (int i = threadIdx.x; i < D; i += 256) { float v = x[i]; s += v; s2 += v * v; }
#pragma unroll
    for (int off = 16; off; off >>= 1) {   // wave32 reduction
        s  += __shfl_down(s, off);
        s2 += __shfl_down(s2, off);
    }
    if ((threadIdx.x & 31) == 0) { sm[threadIdx.x >> 5] = s; sv[threadIdx.x >> 5] = s2; }
    __syncthreads();
    if (threadIdx.x == 0) {
        float ts = 0.0f, ts2 = 0.0f;
#pragma unroll
        for (int i = 0; i < 8; ++i) { ts += sm[i]; ts2 += sv[i]; }
        float mean = ts / (float)D;
        float var  = ts2 / (float)D - mean * mean;
        sm[0] = mean; sv[0] = rsqrtf(var + 1e-5f);
    }
    __syncthreads();
    const float mean = sm[0], rstd = sv[0];
    float* y = Y + (size_t)row * D;
    for (int i = threadIdx.x; i < D; i += 256)
        y[i] = (x[i] - mean) * rstd * g[i] + b[i];
}

// ----------------- causal depthwise conv (k=4) + SiLU ---------------------
// xin = xz[:, 0:DINNER] (ld 2*DINNER); xc[b,l,d] = silu(sum_j w[d,j]*xin[b,l-3+j,d]+bias[d])
__global__ __launch_bounds__(256)
void conv_silu_k(const float* __restrict__ xz, const float* __restrict__ cw,
                 const float* __restrict__ cb, float* __restrict__ xc)
{
    const long long idx = (long long)blockIdx.x * 256 + threadIdx.x;
    if (idx >= (long long)BATCH * SEQL * DINNER) return;
    const int d = (int)(idx % DINNER);
    const int l = (int)((idx / DINNER) % SEQL);
    const int b = (int)(idx / ((long long)DINNER * SEQL));
    float acc = cb[d];
#pragma unroll
    for (int j = 0; j < DCONV; ++j) {
        const int ll = l - (DCONV - 1) + j;
        if (ll >= 0)
            acc += cw[d * DCONV + j] * xz[(size_t)(b * SEQL + ll) * (2 * DINNER) + d];
    }
    const float sg = 1.0f / (1.0f + __expf(-acc));      // SiLU
    xc[(size_t)(b * SEQL + l) * DINNER + d] = acc * sg;
}

// ----------------------- fused selective scan ------------------------------
// One thread per (b,d); 16-state recurrence in registers, fused D-skip and
// silu(z) gate.  xdbc row: [0..63]=dt(unused here), [64..79]=B, [80..95]=C.
__global__ __launch_bounds__(256)
void scan_k(const float* __restrict__ delta, const float* __restrict__ xc,
            const float* __restrict__ xdbc, const float* __restrict__ xz,
            const float* __restrict__ A_log, const float* __restrict__ Dvec,
            float* __restrict__ y)
{
    const int t = blockIdx.x * 256 + threadIdx.x;
    if (t >= BATCH * DINNER) return;
    const int b = t / DINNER;
    const int d = t % DINNER;

    float Ar[DSTATE];
#pragma unroll
    for (int n = 0; n < DSTATE; ++n)
        Ar[n] = -__expf(A_log[d * DSTATE + n]);         // A = -exp(A_log)
    const float Dd = Dvec[d];

    float st[DSTATE];
#pragma unroll
    for (int n = 0; n < DSTATE; ++n) st[n] = 0.0f;

    for (int l = 0; l < SEQL; ++l) {
        const size_t row = (size_t)(b * SEQL + l);
        const float de  = delta[row * DINNER + d];
        const float xcv = xc[row * DINNER + d];
        const float zv  = xz[row * (2 * DINNER) + DINNER + d];
        const float dx  = de * xcv;
        const float* bc = xdbc + row * (DTRANK + 2 * DSTATE);
        float acc = 0.0f;
#pragma unroll
        for (int n = 0; n < DSTATE; ++n) {
            const float dA = __expf(de * Ar[n]);
            st[n] = dA * st[n] + dx * bc[DTRANK + n];          // B_n
            acc  += st[n] * bc[DTRANK + DSTATE + n];           // C_n
        }
        const float yv = acc + xcv * Dd;
        const float zs = zv / (1.0f + __expf(-zv));            // silu(z)
        y[row * DINNER + d] = yv * zs;
    }
}

// --------------------------- launcher --------------------------------------
static inline int gemm_blocks(int M, int N) {
    const int waves = (M >> 5) * ((N + 31) >> 5);
    return (waves + 7) / 8;
}

extern "C" void kernel_launch(void* const* d_in, const int* in_sizes, int n_in,
                              void* d_out, int out_size, void* d_ws, size_t ws_size,
                              hipStream_t stream)
{
    const float* x        = (const float*)d_in[0];   // [B,L,NIN]
    const float* in_w     = (const float*)d_in[1];   // [DMODEL,NIN]
    const float* in_b     = (const float*)d_in[2];   // [DMODEL]
    const float* outp_w   = (const float*)d_in[3];   // [NOUT,DMODEL]
    const float* outp_b   = (const float*)d_in[4];   // [NOUT]
    const float* ln_g     = (const float*)d_in[5];   // [NL,DMODEL]
    const float* ln_b     = (const float*)d_in[6];   // [NL,DMODEL]
    const float* m_in_w   = (const float*)d_in[7];   // [NL,2*DINNER,DMODEL]
    const float* m_conv_w = (const float*)d_in[8];   // [NL,DINNER,1,DCONV]
    const float* m_conv_b = (const float*)d_in[9];   // [NL,DINNER]
    const float* m_xp_w   = (const float*)d_in[10];  // [NL,96,DINNER]
    const float* m_dt_w   = (const float*)d_in[11];  // [NL,DINNER,DTRANK]
    const float* m_dt_b   = (const float*)d_in[12];  // [NL,DINNER]
    const float* m_A_log  = (const float*)d_in[13];  // [NL,DINNER,DSTATE]
    const float* m_D      = (const float*)d_in[14];  // [NL,DINNER]
    const float* m_out_w  = (const float*)d_in[15];  // [NL,DMODEL,DINNER]
    float* out = (float*)d_out;                      // [B,L,NOUT]

    // workspace layout (fp32)
    float* ws    = (float*)d_ws;
    float* h     = ws;                               // [ROWS,DMODEL]
    float* ln    = h     + (size_t)ROWS * DMODEL;    // [ROWS,DMODEL]
    float* xz    = ln    + (size_t)ROWS * DMODEL;    // [ROWS,2*DINNER]
    float* xc    = xz    + (size_t)ROWS * 2 * DINNER;// [ROWS,DINNER]
    float* xdbc  = xc    + (size_t)ROWS * DINNER;    // [ROWS,96]
    float* delta = xdbc  + (size_t)ROWS * (DTRANK + 2 * DSTATE); // [ROWS,DINNER]
    float* ybuf  = delta + (size_t)ROWS * DINNER;    // [ROWS,DINNER]
    float* hnext = ybuf  + (size_t)ROWS * DINNER;    // [ROWS,DMODEL]

    // 1) h = gelu(x @ in_w^T + in_b)
    gemm_bf16_wmma<EPI_BIAS_GELU><<<gemm_blocks(ROWS, DMODEL), 256, 0, stream>>>(
        x, NIN, in_w, NIN, in_b, nullptr, h, DMODEL, ROWS, DMODEL, NIN);

    float* cur = h;
    float* nxt = hnext;
    for (int i = 0; i < NL; ++i) {
        // 2) ln = layernorm(cur)
        layernorm_k<<<ROWS, 256, 0, stream>>>(cur, ln_g + i * DMODEL, ln_b + i * DMODEL, ln, DMODEL);

        // 3) xz = ln @ m_in_w[i]^T   [ROWS, 2*DINNER]
        gemm_bf16_wmma<EPI_NONE><<<gemm_blocks(ROWS, 2 * DINNER), 256, 0, stream>>>(
            ln, DMODEL, m_in_w + (size_t)i * 2 * DINNER * DMODEL, DMODEL,
            nullptr, nullptr, xz, 2 * DINNER, ROWS, 2 * DINNER, DMODEL);

        // 4) xc = silu(causal depthwise conv(xin) + conv_b)
        conv_silu_k<<<(BATCH * SEQL * DINNER + 255) / 256, 256, 0, stream>>>(
            xz, m_conv_w + (size_t)i * DINNER * DCONV, m_conv_b + (size_t)i * DINNER, xc);

        // 5) xdbc = xc @ xproj_w[i]^T   [ROWS, 96]
        gemm_bf16_wmma<EPI_NONE><<<gemm_blocks(ROWS, DTRANK + 2 * DSTATE), 256, 0, stream>>>(
            xc, DINNER, m_xp_w + (size_t)i * (DTRANK + 2 * DSTATE) * DINNER, DINNER,
            nullptr, nullptr, xdbc, DTRANK + 2 * DSTATE, ROWS, DTRANK + 2 * DSTATE, DINNER);

        // 6) delta = softplus(dt @ dt_w[i]^T + dt_b)   (A = xdbc[:, :64], lda=96)
        gemm_bf16_wmma<EPI_BIAS_SOFTPLUS><<<gemm_blocks(ROWS, DINNER), 256, 0, stream>>>(
            xdbc, DTRANK + 2 * DSTATE, m_dt_w + (size_t)i * DINNER * DTRANK, DTRANK,
            m_dt_b + (size_t)i * DINNER, nullptr, delta, DINNER, ROWS, DINNER, DTRANK);

        // 7) fused selective scan + D-skip + silu(z) gate -> ybuf
        scan_k<<<(BATCH * DINNER + 255) / 256, 256, 0, stream>>>(
            delta, xc, xdbc, xz,
            m_A_log + (size_t)i * DINNER * DSTATE, m_D + (size_t)i * DINNER, ybuf);

        // 8) nxt = ybuf @ m_out_w[i]^T + cur   (residual fused)
        gemm_bf16_wmma<EPI_RES><<<gemm_blocks(ROWS, DMODEL), 256, 0, stream>>>(
            ybuf, DINNER, m_out_w + (size_t)i * DMODEL * DINNER, DINNER,
            nullptr, cur, nxt, DMODEL, ROWS, DMODEL, DINNER);

        float* t = cur; cur = nxt; nxt = t;
    }

    // 9) out = cur @ out_proj_w^T + out_proj_b   (N=72, guarded stores only)
    gemm_bf16_wmma<EPI_BIAS><<<gemm_blocks(ROWS, NOUT), 256, 0, stream>>>(
        cur, DMODEL, outp_w, DMODEL, outp_b, nullptr, out, NOUT, ROWS, NOUT, DMODEL);

    (void)in_sizes; (void)n_in; (void)out_size; (void)ws_size;
}
